// SelfAttention_60902636257831
// MI455X (gfx1250) — matrix operand order repaired and checked by reading, not compiled, not run on hardware
//
#include <hip/hip_runtime.h>

#define B_SZ 8
#define C_SZ 512
#define L_SZ 1024
#define H_SZ 8
#define HD   64

typedef __attribute__((ext_vector_type(16))) __bf16 v16bf;
typedef __attribute__((ext_vector_type(8)))  float  v8f;
typedef unsigned int uint32x4 __attribute__((ext_vector_type(4)));
typedef int int32x8 __attribute__((ext_vector_type(8)));
typedef int int32x4 __attribute__((ext_vector_type(4)));

union FragAB { v16bf v; unsigned int u[8]; };

__device__ __forceinline__ unsigned short f2bf(float f) {
  unsigned int x = __float_as_uint(f);
  x += 0x7FFFu + ((x >> 16) & 1u);   // round-to-nearest-even
  return (unsigned short)(x >> 16);
}

// A-matrix 16x32 bf16 lane layout (ISA 7.12.2): pair p, half = lane/16
__device__ __forceinline__ int akoff(int p, int half) {
  return ((p < 4) ? (2 * p) : (16 + 2 * (p - 4))) + 8 * half;
}

// low 32 bits of a flat shared-aperture address == workgroup LDS byte offset
__device__ __forceinline__ unsigned lds_off(const void* p) {
  return (unsigned)(unsigned long long)p;
}

// per-lane async 16B copy global -> LDS (ASYNCcnt), bypasses VGPRs
__device__ __forceinline__ void async_b128(unsigned lds_byte, const void* g) {
  asm volatile("global_load_async_to_lds_b128 %0, %1, off"
               :: "v"(lds_byte), "v"(g) : "memory");
}
__device__ __forceinline__ void wait_async0() {
  asm volatile("s_wait_asynccnt 0x0" ::: "memory");
}

#if __has_builtin(__builtin_amdgcn_tensor_load_to_lds)
#define HAVE_TDM 1
// One-shot TDM tile fetch: rows x (d0_qw * 8B), row stride stride_qw * 8B.
// D# per ISA 8.3/8.4: group0 = {count, lds_addr, global_addr, type=2},
// group1 = {data_size=8B, tensor_dim0/1, tile_dim0/1, dim0_stride}.
// This toolchain's builtin takes 6 args (g0, g1, g2, g3, g4, cpol).
__device__ __forceinline__ void tdm_load_2d(unsigned lds_addr, const void* gptr,
                                            unsigned d0_qw, unsigned rows,
                                            unsigned stride_qw) {
  unsigned long long ga = (unsigned long long)gptr;
  uint32x4 g0;
  g0[0] = 1u;                                                  // count=1 (user D#)
  g0[1] = lds_addr;                                            // LDS byte address
  g0[2] = (unsigned)ga;                                        // global_addr[31:0]
  g0[3] = ((unsigned)(ga >> 32) & 0x01FFFFFFu) | (2u << 30);   // ga[56:32] | type=2
  int32x8 g1;
  g1[0] = (int)(3u << 16);                                     // data_size=3 (8B)
  g1[1] = (int)((d0_qw & 0xFFFFu) << 16);                      // tensor_dim0[15:0]
  g1[2] = (int)(((d0_qw >> 16) & 0xFFFFu) | ((rows & 0xFFFFu) << 16));
  g1[3] = (int)(((rows >> 16) & 0xFFFFu) | ((d0_qw & 0xFFFFu) << 16)); // tile_dim0
  g1[4] = (int)(rows & 0xFFFFu);                               // tile_dim1, tile_dim2=0
  g1[5] = (int)stride_qw;                                      // tensor_dim0_stride
  g1[6] = 0;
  g1[7] = 0;
  int32x4 z4 = {0, 0, 0, 0};
  int32x8 z8 = {0, 0, 0, 0, 0, 0, 0, 0};
  __builtin_amdgcn_tensor_load_to_lds(g0, g1, z4, z4, z8, 0);
}
#else
#define HAVE_TDM 0
#endif

// ---------------- float -> bf16 weight conversion ----------------
__global__ void cvt_bf16_kernel(const float* __restrict__ in,
                                unsigned short* __restrict__ out, int n) {
  int i = blockIdx.x * blockDim.x + threadIdx.x;
  if (i < n) out[i] = f2bf(in[i]);
}

// -------- GroupNorm (8 groups) -> bf16 xnT[b][l][c] (transposed) --------
__global__ __launch_bounds__(256) void gn_bf16_kernel(
    const float* __restrict__ x, const float* __restrict__ gamma,
    const float* __restrict__ beta, unsigned short* __restrict__ xnT) {
  const int b = blockIdx.x >> 3, g = blockIdx.x & 7;
  const size_t base = ((size_t)b * C_SZ + g * 64) * L_SZ;  // 65536 elems
  const float* src = x + base;
  float s = 0.f, s2 = 0.f;
  for (int i = threadIdx.x; i < 64 * L_SZ; i += 256) {
    float v = src[i];
    s += v; s2 += v * v;
  }
#pragma unroll
  for (int m = 1; m < 32; m <<= 1) {
    s  += __shfl_xor(s,  m, 32);
    s2 += __shfl_xor(s2, m, 32);
  }
  __shared__ float red[2][8];
  if ((threadIdx.x & 31) == 0) {
    red[0][threadIdx.x >> 5] = s;
    red[1][threadIdx.x >> 5] = s2;
  }
  __syncthreads();
  s = 0.f; s2 = 0.f;
#pragma unroll
  for (int i = 0; i < 8; i++) { s += red[0][i]; s2 += red[1][i]; }
  const float inv_n = 1.0f / 65536.0f;
  const float mean = s * inv_n;
  const float var  = s2 * inv_n - mean * mean;
  const float rstd = rsqrtf(var + 1e-5f);
  for (int i = threadIdx.x; i < 64 * L_SZ; i += 256) {
    const int c = g * 64 + (i >> 10);
    const int l = i & 1023;
    // transposed store: lands in L2 (42MB ws << 192MB L2), coalesces there
    xnT[((size_t)b * L_SZ + l) * C_SZ + c] =
        f2bf((src[i] - mean) * rstd * gamma[c] + beta[c]);
  }
}

// ------ bf16 WMMA GEMM: D = A(MxK) * B(KxN) + bias, B given transposed ------
// A [M][K], BmT [batch][N][K]  ->  both LDS tiles are straight row copies,
// double-buffered and fetched with global_load_async_to_lds_b128.
// mode 0: scatter q[b][h][l][d], k[b][h][l][d], v[b][h][d][l]   (QKV)
// mode 1: out[b][o][l] = xres[b][o][l] + D                      (proj+resid)
__global__ __launch_bounds__(256) void gemm_wmma_kernel(
    const unsigned short* __restrict__ A,
    const unsigned short* __restrict__ BmT,
    const float* __restrict__ bias, int Kdim, int N, int mode,
    unsigned short* __restrict__ qb, unsigned short* __restrict__ kb,
    unsigned short* __restrict__ vb,
    const float* __restrict__ xres, float* __restrict__ outF) {
  __shared__ __align__(16) unsigned short As[2][128][32];  // [m][k]
  __shared__ __align__(16) unsigned short Bs[2][128][32];  // [n][k]
  const int n0 = blockIdx.x * 128;
  const int m0 = blockIdx.y * 128;
  const int b  = blockIdx.z;
  const int tid = threadIdx.x;
  const int wv = tid >> 5, lane = tid & 31;
  const int half = lane >> 4, rA = lane & 15;
  const v8f vzero = {0.f, 0.f, 0.f, 0.f, 0.f, 0.f, 0.f, 0.f};
  v8f acc[8];
#pragma unroll
  for (int t = 0; t < 8; t++) acc[t] = vzero;

  const int row = tid >> 1, seg = tid & 1;  // 32B per thread per tile
  const unsigned short* gA = A + (size_t)(m0 + row) * Kdim + seg * 16;
  const unsigned short* gB = BmT + ((size_t)b * N + n0 + row) * Kdim + seg * 16;
  const unsigned lA = lds_off(&As[0][row][seg * 16]);
  const unsigned lB = lds_off(&Bs[0][row][seg * 16]);
  const unsigned bufstep = (unsigned)sizeof(As[0]);  // 8KB between buffers

#define STAGE(bf, kk)                                    \
  do {                                                   \
    async_b128(lA + (bf) * bufstep, gA + (kk));          \
    async_b128(lA + (bf) * bufstep + 16, gA + (kk) + 8); \
    async_b128(lB + (bf) * bufstep, gB + (kk));          \
    async_b128(lB + (bf) * bufstep + 16, gB + (kk) + 8); \
  } while (0)

  STAGE(0, 0);
  const int nsteps = Kdim >> 5;
  for (int s = 0; s < nsteps; s++) {
    const int cur = s & 1;
    wait_async0();        // this wave's async copies into buf[cur] done
    __syncthreads();      // everyone's copies visible / prev reads retired
    if (s + 1 < nsteps) STAGE(cur ^ 1, (s + 1) * 32);  // overlap with WMMA
    FragAB fa;
#pragma unroll
    for (int p = 0; p < 8; p++)
      fa.u[p] = *(const unsigned int*)&As[cur][wv * 16 + rA][akoff(p, half)];
#pragma unroll
    for (int t = 0; t < 8; t++) {
      FragAB fb;
#pragma unroll
      for (int p = 0; p < 8; p++)
        fb.u[p] = *(const unsigned int*)&Bs[cur][t * 16 + rA][akoff(p, half)];
      acc[t] = __builtin_amdgcn_wmma_f32_16x16x32_bf16(
          false, fa.v, false, fb.v, (short)0, acc[t], false, false);
    }
    __syncthreads();      // reads done before buf[cur] is overwritten
  }
#undef STAGE

  // epilogue: C layout -> lanes 0-15: (M=r, N=lane); 16-31: (M=r+8, N=lane-16)
#pragma unroll
  for (int t = 0; t < 8; t++) {
    const int l = n0 + t * 16 + rA;
#pragma unroll
    for (int r = 0; r < 8; r++) {
      const int o = m0 + wv * 16 + r + 8 * half;
      float v = acc[t][r] + bias[o];
      if (mode == 0) {
        if (o < 512) {
          const int h = o >> 6, d = o & 63;
          qb[(((size_t)b * H_SZ + h) * L_SZ + l) * HD + d] = f2bf(v);
        } else if (o < 1024) {
          const int oo = o - 512, h = oo >> 6, d = oo & 63;
          kb[(((size_t)b * H_SZ + h) * L_SZ + l) * HD + d] = f2bf(v);
        } else {
          const int oo = o - 1024, h = oo >> 6, d = oo & 63;
          vb[(((size_t)b * H_SZ + h) * HD + d) * L_SZ + l] = f2bf(v);
        }
      } else {
        const size_t idx = ((size_t)b * C_SZ + o) * L_SZ + l;
        outF[idx] = xres[idx] + v;
      }
    }
  }
}

// ---------------- flash attention: per (b,h,128-row q tile) ----------------
__global__ __launch_bounds__(256) void attn_wmma_kernel(
    const unsigned short* __restrict__ qbuf,  // [b][h][l][d]
    const unsigned short* __restrict__ kbuf,  // [b][h][l][d]
    const unsigned short* __restrict__ vbuf,  // [b][h][d][l]
    unsigned short* __restrict__ obT) {       // [b][l][c] transposed
  __shared__ __align__(16) unsigned short Kt[128][64];    // [j][d]
  __shared__ __align__(16) unsigned short Vt[64][128];    // [d][j]
  __shared__ __align__(16) unsigned short Pt[8][16][128]; // wave-private P
  const int q0 = blockIdx.x * 128;
  const int h = blockIdx.y, b = blockIdx.z;
  const int tid = threadIdx.x, wv = tid >> 5, lane = tid & 31;
  const int half = lane >> 4, rA = lane & 15;
  const size_t bh = (size_t)b * H_SZ + h;

  FragAB fq[2];  // Q rows pinned in registers for the whole kernel
  {
    const unsigned short* qrow = qbuf + (bh * L_SZ + q0 + wv * 16 + rA) * HD;
#pragma unroll
    for (int ks = 0; ks < 2; ks++)
#pragma unroll
      for (int p = 0; p < 8; p++)
        fq[ks].u[p] = *(const unsigned int*)(qrow + akoff(p, half) + ks * 32);
  }
  const v8f vzero = {0.f, 0.f, 0.f, 0.f, 0.f, 0.f, 0.f, 0.f};
  v8f oAcc[4];
#pragma unroll
  for (int t = 0; t < 4; t++) oAcc[t] = vzero;
  float mrow[8], lrow[8];
#pragma unroll
  for (int r = 0; r < 8; r++) { mrow[r] = -1e30f; lrow[r] = 0.f; }

  for (int jt = 0; jt < 8; jt++) {
    const int j0 = jt * 128;
    __syncthreads();  // previous iteration's reads of Kt/Vt retired
#if HAVE_TDM
    if (wv == 0) {
      // K tile: contiguous 16KB (2048 x 8B), one TDM op
      tdm_load_2d(lds_off(&Kt[0][0]), kbuf + (bh * L_SZ + j0) * HD, 2048, 1, 2048);
      // V tile: 64 rows x 256B, row stride 2048B
      tdm_load_2d(lds_off(&Vt[0][0]), vbuf + bh * HD * L_SZ + j0, 32, 64, 256);
      __builtin_amdgcn_s_wait_tensorcnt(0);
    }
#else
    {
      const uint4* s = (const uint4*)(kbuf + (bh * L_SZ + j0) * HD);
      uint4* d = (uint4*)&Kt[0][0];
#pragma unroll
      for (int i = 0; i < 4; i++) d[tid + 256 * i] = s[tid + 256 * i];
      const int dd = tid >> 2, sg = tid & 3;
      const uint4* sv = (const uint4*)(vbuf + (bh * HD + dd) * L_SZ + j0 + sg * 32);
      uint4* dv = (uint4*)&Vt[dd][sg * 32];
      dv[0] = sv[0]; dv[1] = sv[1]; dv[2] = sv[2]; dv[3] = sv[3];
    }
#endif
    __syncthreads();  // tiles visible to all waves
    // S = (Q K^T) * scale
    v8f sAcc[8];
#pragma unroll
    for (int t = 0; t < 8; t++) sAcc[t] = vzero;
#pragma unroll
    for (int ks = 0; ks < 2; ks++) {
#pragma unroll
      for (int t = 0; t < 8; t++) {
        FragAB fb;
#pragma unroll
        for (int p = 0; p < 8; p++)
          fb.u[p] = *(const unsigned int*)&Kt[t * 16 + rA][akoff(p, half) + ks * 32];
        sAcc[t] = __builtin_amdgcn_wmma_f32_16x16x32_bf16(
            false, fq[ks].v, false, fb.v, (short)0, sAcc[t], false, false);
      }
    }
#pragma unroll
    for (int t = 0; t < 8; t++)
#pragma unroll
      for (int r = 0; r < 8; r++) sAcc[t][r] *= 0.125f;  // 1/sqrt(64)
    // online softmax: rows live in 16-lane halves -> shfl_xor 1,2,4,8
    float mnew[8], resc[8], rs[8];
#pragma unroll
    for (int r = 0; r < 8; r++) {
      float tm = sAcc[0][r];
#pragma unroll
      for (int t = 1; t < 8; t++) tm = fmaxf(tm, sAcc[t][r]);
#pragma unroll
      for (int msk = 1; msk < 16; msk <<= 1) tm = fmaxf(tm, __shfl_xor(tm, msk, 32));
      mnew[r] = fmaxf(mrow[r], tm);
      resc[r] = __expf(mrow[r] - mnew[r]);
      mrow[r] = mnew[r];
      rs[r] = 0.f;
    }
#pragma unroll
    for (int t = 0; t < 8; t++)
#pragma unroll
      for (int r = 0; r < 8; r++) {
        float e = __expf(sAcc[t][r] - mnew[r]);
        rs[r] += e;
        Pt[wv][r + 8 * half][t * 16 + rA] = f2bf(e);  // wave-private
      }
#pragma unroll
    for (int r = 0; r < 8; r++) {
      float t2 = rs[r];
#pragma unroll
      for (int msk = 1; msk < 16; msk <<= 1) t2 += __shfl_xor(t2, msk, 32);
      lrow[r] = lrow[r] * resc[r] + t2;
    }
#pragma unroll
    for (int t = 0; t < 4; t++)
#pragma unroll
      for (int r = 0; r < 8; r++) oAcc[t][r] *= resc[r];
    // O += P V   (Pt wave-private: same-wave DS ordering suffices)
#pragma unroll
    for (int ks = 0; ks < 4; ks++) {
      FragAB fp;
#pragma unroll
      for (int p = 0; p < 8; p++)
        fp.u[p] = *(const unsigned int*)&Pt[wv][rA][akoff(p, half) + ks * 32];
#pragma unroll
      for (int t = 0; t < 4; t++) {
        FragAB fv;
#pragma unroll
        for (int p = 0; p < 8; p++)
          fv.u[p] = *(const unsigned int*)&Vt[t * 16 + rA][akoff(p, half) + ks * 32];
        oAcc[t] = __builtin_amdgcn_wmma_f32_16x16x32_bf16(
            false, fp.v, false, fv.v, (short)0, oAcc[t], false, false);
      }
    }
  }
  // normalize; write transposed obT[b][l][h*64+d] so proj GEMM stages flat
#pragma unroll
  for (int t = 0; t < 4; t++) {
    const int d = t * 16 + rA;
#pragma unroll
    for (int r = 0; r < 8; r++) {
      const int l = q0 + wv * 16 + r + 8 * half;
      obT[((size_t)b * L_SZ + l) * C_SZ + h * HD + d] = f2bf(oAcc[t][r] / lrow[r]);
    }
  }
}

extern "C" void kernel_launch(void* const* d_in, const int* in_sizes, int n_in,
                              void* d_out, int out_size, void* d_ws, size_t ws_size,
                              hipStream_t stream) {
  (void)in_sizes; (void)n_in; (void)out_size; (void)ws_size;
  const float* x     = (const float*)d_in[0];
  const float* gamma = (const float*)d_in[1];
  const float* beta  = (const float*)d_in[2];
  const float* wqkv  = (const float*)d_in[3];
  const float* bqkv  = (const float*)d_in[4];
  const float* wproj = (const float*)d_in[5];
  const float* bproj = (const float*)d_in[6];
  float* out = (float*)d_out;
  char* ws = (char*)d_ws;
  unsigned short* xnT    = (unsigned short*)(ws + 0);         // 8 MB  [b][l][c]
  unsigned short* wqkvb  = (unsigned short*)(ws + 8388608);   // 1.5MB
  unsigned short* wprojb = (unsigned short*)(ws + 9961472);   // 0.5MB
  unsigned short* qb     = (unsigned short*)(ws + 10485760);  // 8 MB [b][h][l][d]
  unsigned short* kb     = (unsigned short*)(ws + 18874368);  // 8 MB [b][h][l][d]
  unsigned short* vb     = (unsigned short*)(ws + 27262976);  // 8 MB [b][h][d][l]
  unsigned short* obT    = (unsigned short*)(ws + 35651584);  // 8 MB [b][l][c]

  cvt_bf16_kernel<<<(1536 * 512 + 255) / 256, 256, 0, stream>>>(wqkv, wqkvb, 1536 * 512);
  cvt_bf16_kernel<<<(512 * 512 + 255) / 256, 256, 0, stream>>>(wproj, wprojb, 512 * 512);
  gn_bf16_kernel<<<64, 256, 0, stream>>>(x, gamma, beta, xnT);
  // QKV: M=1536, N=1024, K=512 per batch
  gemm_wmma_kernel<<<dim3(8, 12, 8), 256, 0, stream>>>(
      wqkvb, xnT, bqkv, 512, 1024, 0, qb, kb, vb, nullptr, nullptr);
  attn_wmma_kernel<<<dim3(8, 8, 8), 256, 0, stream>>>(qb, kb, vb, obT);
  // proj + residual: M=512, N=1024, K=512 per batch
  gemm_wmma_kernel<<<dim3(8, 4, 8), 256, 0, stream>>>(
      wprojb, obT, bproj, 512, 1024, 1, nullptr, nullptr, nullptr, x, out);
}
